// PAGAGCN_29128468201963
// MI455X (gfx1250) — matrix operand order repaired and checked by reading, not compiled, not run on hardware
//
#include <hip/hip_runtime.h>

typedef __bf16 bf16_t;
typedef __attribute__((ext_vector_type(16))) __bf16 v16bf;
typedef __attribute__((ext_vector_type(8)))  float  v8f;

#define NN 1024
#define NS (NN * NN)
#define NE 32768

static inline int ceil_div(int a, int b) { return (a + b - 1) / b; }

// ---------------------------------------------------------------------------
// WMMA GEMM: C[M x N] = A[M x K] * B[K x N], with B supplied transposed
// (Bt[n*K + k] = B[k*N + n]).  bf16 inputs, f32 accumulate.
// mode: 0 = plain store, 1 = +bias[col] then ReLU, 2 = +bias[col]
// Block = 128 threads (4 waves), block tile 64x64, wave tile 32x32.
// ---------------------------------------------------------------------------
union FragAB {
    v16bf v;
    uint4 q[2];
};

__global__ void __launch_bounds__(128)
wmma_gemm_bf16_k(const bf16_t* __restrict__ A, const bf16_t* __restrict__ Bt,
                 float* __restrict__ C, int M, int Nn, int K,
                 const float* __restrict__ bias, int mode)
{
    const int lane = threadIdx.x & 31;
    const int wave = threadIdx.x >> 5;        // 0..3
    const int wm = wave >> 1, wn = wave & 1;  // 2x2 wave grid
    const int row0 = blockIdx.y * 64 + wm * 32;
    const int col0 = blockIdx.x * 64 + wn * 32;
    if (row0 >= M || col0 >= Nn) return;      // wave-uniform exit, EXEC stays full

    const int mh   = lane & 15;               // M (or N) index within 16
    const int asel = (lane >> 4) & 1;         // K-half select per ISA layout

    // A fragment (16x32 bf16): lanes 0-15 hold K = {0..7,16..23}, lanes 16-31
    // hold K = {8..15,24..31}: two contiguous 16B chunks per lane.
    const bf16_t* Ar0 = A + (size_t)(row0 + mh) * K + asel * 8;
    const bf16_t* Ar1 = A + (size_t)(row0 + 16 + mh) * K + asel * 8;
    // B fragment (32x16 bf16) from transposed B: lanes 0-15 hold K = {0..7,16..23},
    // lanes 16-31 hold K = {8..15,24..31}: two contiguous 16B chunks per lane.
    const bf16_t* Bc0 = Bt + (size_t)(col0 + mh) * K + asel * 8;
    const bf16_t* Bc1 = Bt + (size_t)(col0 + 16 + mh) * K + asel * 8;

    v8f acc00 = {}, acc01 = {}, acc10 = {}, acc11 = {};

    for (int k0 = 0; k0 < K; k0 += 32) {
        FragAB a0, a1, b0, b1;
        a0.q[0] = *reinterpret_cast<const uint4*>(Ar0 + k0);
        a0.q[1] = *reinterpret_cast<const uint4*>(Ar0 + k0 + 16);
        a1.q[0] = *reinterpret_cast<const uint4*>(Ar1 + k0);
        a1.q[1] = *reinterpret_cast<const uint4*>(Ar1 + k0 + 16);
        b0.q[0] = *reinterpret_cast<const uint4*>(Bc0 + k0);
        b0.q[1] = *reinterpret_cast<const uint4*>(Bc0 + k0 + 16);
        b1.q[0] = *reinterpret_cast<const uint4*>(Bc1 + k0);
        b1.q[1] = *reinterpret_cast<const uint4*>(Bc1 + k0 + 16);

        __builtin_prefetch(Ar0 + k0 + 64, 0, 3);   // global_prefetch_b8
        __builtin_prefetch(Bc0 + k0 + 64, 0, 3);

        acc00 = __builtin_amdgcn_wmma_f32_16x16x32_bf16(false, a0.v, false, b0.v,
                                                        (short)0, acc00, false, false);
        acc01 = __builtin_amdgcn_wmma_f32_16x16x32_bf16(false, a0.v, false, b1.v,
                                                        (short)0, acc01, false, false);
        acc10 = __builtin_amdgcn_wmma_f32_16x16x32_bf16(false, a1.v, false, b0.v,
                                                        (short)0, acc10, false, false);
        acc11 = __builtin_amdgcn_wmma_f32_16x16x32_bf16(false, a1.v, false, b1.v,
                                                        (short)0, acc11, false, false);
    }

    // C/D layout: lane = N index (mh), VGPR v = row (v + asel*8) within 16-tile.
    const int c0 = col0 + mh;
    const int c1 = col0 + 16 + mh;
    float bv0 = 0.f, bv1 = 0.f;
    if (mode != 0) { bv0 = bias[c0]; bv1 = bias[c1]; }

#pragma unroll
    for (int v = 0; v < 8; ++v) {
        const int r0 = row0 + asel * 8 + v;
        const int r1 = row0 + 16 + asel * 8 + v;
        float x00 = acc00[v] + bv0, x01 = acc01[v] + bv1;
        float x10 = acc10[v] + bv0, x11 = acc11[v] + bv1;
        if (mode == 1) {
            x00 = fmaxf(x00, 0.f); x01 = fmaxf(x01, 0.f);
            x10 = fmaxf(x10, 0.f); x11 = fmaxf(x11, 0.f);
        }
        C[(size_t)r0 * Nn + c0] = x00;
        C[(size_t)r0 * Nn + c1] = x01;
        C[(size_t)r1 * Nn + c0] = x10;
        C[(size_t)r1 * Nn + c1] = x11;
    }
}

// ---------------------------------------------------------------------------
// Elementwise / packing helpers
// ---------------------------------------------------------------------------
__global__ void fill_zero_k(float* p, int n)
{
    int i = blockIdx.x * blockDim.x + threadIdx.x;
    if (i < n) p[i] = 0.f;
}

__global__ void copy_k(const float* __restrict__ s, float* __restrict__ d, int n)
{
    int i = blockIdx.x * blockDim.x + threadIdx.x;
    if (i < n) d[i] = s[i];
}

__global__ void max_k(const float* __restrict__ a, const float* __restrict__ b,
                      float* __restrict__ d, int n)
{
    int i = blockIdx.x * blockDim.x + threadIdx.x;
    if (i < n) d[i] = fmaxf(a[i], b[i]);
}

__global__ void f32_to_bf16_k(const float* __restrict__ s, bf16_t* __restrict__ d, int n)
{
    int i = blockIdx.x * blockDim.x + threadIdx.x;
    if (i < n) d[i] = (bf16_t)s[i];
}

// src is K x Nn row-major; dst is Nn x K (transposed pack)
__global__ void f32_to_bf16_T_k(const float* __restrict__ s, bf16_t* __restrict__ d,
                                int K, int Nn)
{
    int i = blockIdx.x * blockDim.x + threadIdx.x;
    if (i < K * Nn) {
        int n = i / K, k = i % K;
        d[i] = (bf16_t)s[(size_t)k * Nn + n];
    }
}

__global__ void scatter_edges_k(const int* __restrict__ e, float* __restrict__ A, int ne)
{
    int i = blockIdx.x * blockDim.x + threadIdx.x;
    if (i < ne) {
        int r = e[i], c = e[ne + i];
        atomicAdd(&A[(size_t)r * NN + c], 1.0f);
    }
}

// ---------------------------------------------------------------------------
// dense GCN norm: add self loops where diag==0, dinv = rsqrt(rowsum),
// A <- dinv[i] * A * dinv[j]
// ---------------------------------------------------------------------------
__global__ void fix_diag_k(float* A, int n)
{
    int i = blockIdx.x * blockDim.x + threadIdx.x;
    if (i < n && A[(size_t)i * n + i] == 0.f) A[(size_t)i * n + i] = 1.f;
}

__global__ void row_dinv_k(const float* __restrict__ A, float* __restrict__ dinv, int n)
{
    __shared__ float red[256];
    float s = 0.f;
    const float* row = A + (size_t)blockIdx.x * n;
    for (int j = threadIdx.x; j < n; j += 256) s += row[j];
    red[threadIdx.x] = s;
    __syncthreads();
    for (int off = 128; off > 0; off >>= 1) {
        if (threadIdx.x < off) red[threadIdx.x] += red[threadIdx.x + off];
        __syncthreads();
    }
    if (threadIdx.x == 0) {
        float d = red[0];
        dinv[blockIdx.x] = (d > 0.f) ? rsqrtf(d) : 0.f;
    }
}

__global__ void scale_sym_k(float* __restrict__ A, const float* __restrict__ dinv, int n)
{
    int idx = blockIdx.x * blockDim.x + threadIdx.x;
    if (idx < n * n) {
        int i = idx / n, j = idx % n;
        A[idx] *= dinv[i] * dinv[j];
    }
}

// ---------------------------------------------------------------------------
// Attention conv encoder (fp32 VALU; small fraction of total FLOPs)
// ---------------------------------------------------------------------------
// conv3x3(pad1, L->16) + ReLU + maxpool2 : in L x1024x1024 -> out 16x512x512
__global__ void conv1_relu_pool_k(const float* __restrict__ in, const float* __restrict__ w,
                                  const float* __restrict__ b, float* __restrict__ out, int L)
{
    const int HO = 512, HI = 1024;
    int idx = blockIdx.x * blockDim.x + threadIdx.x;
    if (idx >= 16 * HO * HO) return;
    int ox = idx % HO, oy = (idx / HO) % HO, co = idx / (HO * HO);
    float m = -1e30f;
    for (int dy = 0; dy < 2; ++dy)
        for (int dx = 0; dx < 2; ++dx) {
            int y = 2 * oy + dy, x = 2 * ox + dx;
            float acc = b[co];
            for (int ci = 0; ci < L; ++ci)
                for (int ky = 0; ky < 3; ++ky) {
                    int yy = y - 1 + ky;
                    if (yy < 0 || yy >= HI) continue;
                    for (int kx = 0; kx < 3; ++kx) {
                        int xx = x - 1 + kx;
                        if (xx < 0 || xx >= HI) continue;
                        acc += in[((size_t)ci * HI + yy) * HI + xx] *
                               w[((co * L + ci) * 3 + ky) * 3 + kx];
                    }
                }
            m = fmaxf(m, fmaxf(acc, 0.f));
        }
    out[idx] = m;
}

// conv3x3(pad2, 16->4) + ReLU + maxpool2 : 16x512x512 -> conv 514 -> pool 4x257x257
__global__ void conv2_relu_pool_k(const float* __restrict__ in, const float* __restrict__ w,
                                  const float* __restrict__ b, float* __restrict__ out)
{
    const int HO = 257, HI = 512;
    int idx = blockIdx.x * blockDim.x + threadIdx.x;
    if (idx >= 4 * HO * HO) return;
    int ox = idx % HO, oy = (idx / HO) % HO, co = idx / (HO * HO);
    float m = -1e30f;
    for (int dy = 0; dy < 2; ++dy)
        for (int dx = 0; dx < 2; ++dx) {
            int y = 2 * oy + dy, x = 2 * ox + dx;   // < 514
            float acc = b[co];
            for (int ci = 0; ci < 16; ++ci)
                for (int ky = 0; ky < 3; ++ky) {
                    int yy = y - 2 + ky;
                    if (yy < 0 || yy >= HI) continue;
                    for (int kx = 0; kx < 3; ++kx) {
                        int xx = x - 2 + kx;
                        if (xx < 0 || xx >= HI) continue;
                        acc += in[((size_t)ci * HI + yy) * HI + xx] *
                               w[((co * 16 + ci) * 3 + ky) * 3 + kx];
                    }
                }
            m = fmaxf(m, fmaxf(acc, 0.f));
        }
    out[idx] = m;
}

// tconv2x2 stride2 (4->16) + ReLU : 4x257x257 -> 16x514x514
__global__ void tconv1_relu_k(const float* __restrict__ in, const float* __restrict__ w,
                              const float* __restrict__ b, float* __restrict__ out)
{
    const int HO = 514, HI = 257;
    int idx = blockIdx.x * blockDim.x + threadIdx.x;
    if (idx >= 16 * HO * HO) return;
    int ox = idx % HO, oy = (idx / HO) % HO, co = idx / (HO * HO);
    int ky = 1 - (oy & 1), iy = (oy - 1 + ky) >> 1;
    int kx = 1 - (ox & 1), ix = (ox - 1 + kx) >> 1;
    float acc = b[co];
    for (int ci = 0; ci < 4; ++ci)
        acc += in[((size_t)ci * HI + iy) * HI + ix] * w[((co * 4 + ci) * 2 + ky) * 2 + kx];
    out[idx] = fmaxf(acc, 0.f);
}

// tconv2x2 stride2 (16->L) + sigmoid, fused with attended = adj * att (crop 1024)
__global__ void tconv2_sig_mul_k(const float* __restrict__ t1, const float* __restrict__ w,
                                 const float* __restrict__ b, const float* __restrict__ adj,
                                 float* __restrict__ attended, int L)
{
    const int HI = 514;
    int idx = blockIdx.x * blockDim.x + threadIdx.x;
    if (idx >= L * NS) return;
    int x = idx % NN, y = (idx / NN) % NN, c = idx / NS;
    int ky = 1 - (y & 1), iy = (y - 1 + ky) >> 1;
    int kx = 1 - (x & 1), ix = (x - 1 + kx) >> 1;
    float acc = b[c];
    for (int ci = 0; ci < 16; ++ci)
        acc += t1[((size_t)ci * HI + iy) * HI + ix] * w[((c * 16 + ci) * 2 + ky) * 2 + kx];
    float s = 1.f / (1.f + __expf(-acc));
    attended[idx] = adj[idx] * s;
}

// ---------------------------------------------------------------------------
// Driver
// ---------------------------------------------------------------------------
extern "C" void kernel_launch(void* const* d_in, const int* in_sizes, int n_in,
                              void* d_out, int out_size, void* d_ws, size_t ws_size,
                              hipStream_t stream)
{
    (void)in_sizes; (void)n_in; (void)out_size; (void)ws_size;

    const float* x     = (const float*)d_in[0];
    const int*   e[5]  = { (const int*)d_in[1], (const int*)d_in[2], (const int*)d_in[3],
                           (const int*)d_in[4], (const int*)d_in[5] };
    const float* w1    = (const float*)d_in[22];
    const float* bias1 = (const float*)d_in[23];
    const float* w2    = (const float*)d_in[24];
    const float* bias2 = (const float*)d_in[25];

    // carve workspace
    size_t off = 0;
    auto carve = [&](size_t bytes) -> void* {
        void* p = (char*)d_ws + off;
        off += (bytes + 255) & ~(size_t)255;
        return p;
    };
    float*  adj  = (float*)carve((size_t)5 * NS * sizeof(float));
    float*  att  = (float*)carve((size_t)5 * NS * sizeof(float));
    float*  p1   = (float*)carve((size_t)16 * 512 * 512 * sizeof(float));
    float*  p2   = (float*)carve((size_t)4 * 257 * 257 * sizeof(float));
    float*  t1   = (float*)carve((size_t)16 * 514 * 514 * sizeof(float));
    float*  Ma   = (float*)carve((size_t)NS * sizeof(float));
    float*  Mb   = (float*)carve((size_t)NS * sizeof(float));
    float*  M0   = (float*)carve((size_t)NS * sizeof(float));
    float*  M1   = (float*)carve((size_t)NS * sizeof(float));
    float*  dinv = (float*)carve((size_t)NN * sizeof(float));
    bf16_t* Abf  = (bf16_t*)carve((size_t)NS * sizeof(bf16_t));
    bf16_t* Btbf = (bf16_t*)carve((size_t)NS * sizeof(bf16_t));
    bf16_t* Anbf = (bf16_t*)carve((size_t)NS * sizeof(bf16_t));
    float*  XW1  = (float*)carve((size_t)NN * 64 * sizeof(float));
    float*  Hbuf = (float*)carve((size_t)NN * 64 * sizeof(float));
    float*  HW2  = (float*)carve((size_t)NN * 32 * sizeof(float));

    const int T = 256;
    auto run_norm = [&](float* A) {
        fix_diag_k<<<ceil_div(NN, T), T, 0, stream>>>(A, NN);
        row_dinv_k<<<NN, T, 0, stream>>>(A, dinv, NN);
        scale_sym_k<<<ceil_div(NS, T), T, 0, stream>>>(A, dinv, NN);
    };
    auto run_gemm = [&](const bf16_t* A_, const bf16_t* B_, float* C_, int M_, int N_,
                        int K_, const float* bias_, int mode_) {
        dim3 g(ceil_div(N_, 64), ceil_div(M_, 64));
        wmma_gemm_bf16_k<<<g, 128, 0, stream>>>(A_, B_, C_, M_, N_, K_, bias_, mode_);
    };
    auto to_bf = [&](const float* s, bf16_t* d, int n) {
        f32_to_bf16_k<<<ceil_div(n, T), T, 0, stream>>>(s, d, n);
    };
    auto to_bfT = [&](const float* s, bf16_t* d, int K_, int N_) {
        f32_to_bf16_T_k<<<ceil_div(K_ * N_, T), T, 0, stream>>>(s, d, K_, N_);
    };

    // 1) dense adjacencies (sum duplicate edges)
    fill_zero_k<<<ceil_div(5 * NS, T), T, 0, stream>>>(adj, 5 * NS);
    for (int i = 0; i < 5; ++i)
        scatter_edges_k<<<ceil_div(NE, T), T, 0, stream>>>(e[i], adj + (size_t)i * NS, NE);

    // 2) attention encoders -> attended = adj * sigmoid(att)
    auto run_block = [&](const float* adjb, int L, int pbase, float* attb) {
        const float* c1w = (const float*)d_in[pbase + 0];
        const float* c1b = (const float*)d_in[pbase + 1];
        const float* c2w = (const float*)d_in[pbase + 2];
        const float* c2b = (const float*)d_in[pbase + 3];
        const float* t1w = (const float*)d_in[pbase + 4];
        const float* t1b = (const float*)d_in[pbase + 5];
        const float* t2w = (const float*)d_in[pbase + 6];
        const float* t2b = (const float*)d_in[pbase + 7];
        conv1_relu_pool_k<<<ceil_div(16 * 512 * 512, T), T, 0, stream>>>(adjb, c1w, c1b, p1, L);
        conv2_relu_pool_k<<<ceil_div(4 * 257 * 257, T), T, 0, stream>>>(p1, c2w, c2b, p2);
        tconv1_relu_k<<<ceil_div(16 * 514 * 514, T), T, 0, stream>>>(p2, t1w, t1b, t1);
        tconv2_sig_mul_k<<<ceil_div(L * NS, T), T, 0, stream>>>(t1, t2w, t2b, adjb, attb, L);
    };
    run_block(adj, 2, 6, att);                              // meta-path 0 (L=2)
    run_block(adj + (size_t)2 * NS, 3, 14, att + (size_t)2 * NS);  // meta-path 1 (L=3)

    // 3) path transform: M = gcn_norm(M @ attended[i]) chain (hop 0: I @ A0 = A0)
    auto run_path = [&](const float* attb, int L, float* outM) {
        copy_k<<<ceil_div(NS, T), T, 0, stream>>>(attb, Ma, NS);
        run_norm(Ma);
        float* cur = Ma; float* nxt = Mb;
        for (int h = 1; h < L; ++h) {
            to_bf(cur, Abf, NS);
            to_bfT(attb + (size_t)h * NS, Btbf, NN, NN);
            run_gemm(Abf, Btbf, nxt, NN, NN, NN, nullptr, 0);
            run_norm(nxt);
            float* t = cur; cur = nxt; nxt = t;
        }
        copy_k<<<ceil_div(NS, T), T, 0, stream>>>(cur, outM, NS);
    };
    run_path(att, 2, M0);
    run_path(att + (size_t)2 * NS, 3, M1);

    // 4) A = max(M0, M1); An = gcn_norm(A)
    max_k<<<ceil_div(NS, T), T, 0, stream>>>(M0, M1, Ma, NS);
    run_norm(Ma);                               // An = Ma
    to_bf(Ma, Anbf, NS);

    // 5) GCN layers: h = relu(An @ (x@w1) + bias1); out = An @ (h@w2) + bias2
    to_bf(x, Abf, NN * 128);
    to_bfT(w1, Btbf, 128, 64);
    run_gemm(Abf, Btbf, XW1, NN, 64, 128, nullptr, 0);

    to_bfT(XW1, Btbf, NN, 64);
    run_gemm(Anbf, Btbf, Hbuf, NN, 64, NN, bias1, 1);       // +bias1, ReLU

    to_bf(Hbuf, Abf, NN * 64);
    to_bfT(w2, Btbf, 64, 32);
    run_gemm(Abf, Btbf, HW2, NN, 32, 64, nullptr, 0);

    to_bfT(HW2, Btbf, NN, 32);
    run_gemm(Anbf, Btbf, (float*)d_out, NN, 32, NN, bias2, 2);  // +bias2
}
